// GraphAttentionLayer_44186623542148
// MI455X (gfx1250) — matrix operand order repaired and checked by reading, not compiled, not run on hardware
//
#include <hip/hip_runtime.h>
#include <hip/hip_bf16.h>

typedef __attribute__((ext_vector_type(16))) _Float16 v16h;
typedef __attribute__((ext_vector_type(8)))  _Float16 v8h;
typedef __attribute__((ext_vector_type(8)))  float    v8f;
typedef __attribute__((ext_vector_type(4)))  float    v4f;

#define WMMA_F16(a, b, c) \
  __builtin_amdgcn_wmma_f32_16x16x32_f16(false, (a), false, (b), (short)0, (c), false, false)

#define Bdim 8
#define Ndim 512
#define Ddim 128
#define Hdim 8
#define Edim 16

// ---------------- reductions (wave32) ----------------
__device__ __forceinline__ float wave_sum32(float x) {
#pragma unroll
  for (int o = 16; o >= 1; o >>= 1) x += __shfl_xor(x, o, 32);
  return x;
}
__device__ __forceinline__ float half_max16(float x) {
#pragma unroll
  for (int o = 8; o >= 1; o >>= 1) x = fmaxf(x, __shfl_xor(x, o, 32));
  return x;
}
__device__ __forceinline__ float half_sum16(float x) {
#pragma unroll
  for (int o = 8; o >= 1; o >>= 1) x += __shfl_xor(x, o, 32);
  return x;
}

// ---------------- WMMA operand loaders ----------------
// A 16x32 f16: lane holds row (lane&15); element e -> k = e+16*half
__device__ __forceinline__ v16h ldsA(const _Float16* src, int ld, int k0, int lane) {
  const int row = lane & 15, half = lane >> 4;
  v16h a;
#pragma unroll
  for (int e = 0; e < 16; ++e) {
    const int k = k0 + e + 16 * half;
    a[e] = src[row * ld + k];
  }
  return a;
}
// A with only K=16 valid (zero-padded to 32)
__device__ __forceinline__ v16h ldsA_pad16(const _Float16* src, int ld, int lane) {
  const int row = lane & 15, half = lane >> 4;
  v16h a;
#pragma unroll
  for (int e = 0; e < 16; ++e)
    a[e] = (half == 0) ? src[row * ld + e] : (_Float16)0.f;
  return a;
}
// B operand from TRANSPOSED f16 weights Wt[n][k]: one contiguous 32B load per lane
__device__ __forceinline__ v16h gBt(const _Float16* __restrict__ Wt, int K, int k0, int n0, int lane) {
  return *(const v16h*)(Wt + (size_t)(n0 + (lane & 15)) * K + k0 + 16 * (lane >> 4));
}

// ======== kernel 0: one-time f32 -> f16 TRANSPOSED weight packing ========
__global__ __launch_bounds__(256) void prep_kernel(
    const float* __restrict__ Wp, const float* __restrict__ Wq,
    const float* __restrict__ Wk, const float* __restrict__ Wv,
    const float* __restrict__ ffW1, const float* __restrict__ ffW2,
    const float* __restrict__ We1, const float* __restrict__ We2,
    _Float16* __restrict__ wpt, _Float16* __restrict__ wqt,
    _Float16* __restrict__ wkt, _Float16* __restrict__ wvt,
    _Float16* __restrict__ w1t, _Float16* __restrict__ w2t,
    _Float16* __restrict__ we1t, _Float16* __restrict__ we2t) {
  const int t = blockIdx.x * 256 + threadIdx.x;   // grid covers 32768
  if (t < 16384) {                                 // [128][128] -> Wt[n*128+k]
    const int k = t >> 7, n = t & 127;
    wpt[n * 128 + k] = (_Float16)Wp[t];
    wqt[n * 128 + k] = (_Float16)Wq[t];
    wkt[n * 128 + k] = (_Float16)Wk[t];
    wvt[n * 128 + k] = (_Float16)Wv[t];
  }
  if (t < 32768) {                                 // ffW1 [128][256] -> w1t[n*128+k]
    const int k = t >> 8, n = t & 255;
    w1t[n * 128 + k] = (_Float16)ffW1[t];
  }
  if (t < 32768) {                                 // ffW2 [256][128] -> w2t[n*256+k]
    const int k = t >> 7, n = t & 127;
    w2t[n * 256 + k] = (_Float16)ffW2[t];
  }
  if (t < 512) {                                   // We1 [16][32] -> we1t[n*16+k]
    const int k = t >> 5, n = t & 31;
    we1t[n * 16 + k] = (_Float16)We1[t];
  }
  if (t < 512) {                                   // We2 [32][8] -> padded we2t[16][32]
    const int n = t >> 5, k = t & 31;
    we2t[t] = (n < 8) ? (_Float16)We2[k * 8 + n] : (_Float16)0.f;
  }
}

// =========== kernel 1: LN1 -> xp -> Q(f16,pre-scaled), K(f16), V(f16 transposed) ===========
__global__ __launch_bounds__(32) void node_proj_kernel(
    const float* __restrict__ x, const float* __restrict__ ln1_g, const float* __restrict__ ln1_b,
    const float* __restrict__ bp,
    const _Float16* __restrict__ wpt, const _Float16* __restrict__ wqt,
    const _Float16* __restrict__ wkt, const _Float16* __restrict__ wvt,
    float* __restrict__ xp, _Float16* __restrict__ qh, _Float16* __restrict__ kh,
    _Float16* __restrict__ vth) {
  __shared__ __align__(32) _Float16 aL[16 * 128];
  __shared__ __align__(32) _Float16 xpL[16 * 128];
  const int lane = threadIdx.x;
  const int row0 = blockIdx.x * 16;
  const int half = lane >> 4, cl = lane & 15;
#pragma unroll
  for (int r = 0; r < 16; ++r) {
    const float* xr = x + (size_t)(row0 + r) * 128;
    float vals[4], s = 0.f, s2 = 0.f;
#pragma unroll
    for (int c = 0; c < 4; ++c) { float t = xr[lane + 32 * c]; vals[c] = t; s += t; s2 += t * t; }
    s = wave_sum32(s); s2 = wave_sum32(s2);
    const float mean = s * (1.f / 128.f);
    const float var  = s2 * (1.f / 128.f) - mean * mean;
    const float rinv = rsqrtf(var + 1e-5f);
#pragma unroll
    for (int c = 0; c < 4; ++c) {
      const int col = lane + 32 * c;
      aL[r * 128 + col] = (_Float16)((vals[c] - mean) * rinv * ln1_g[col] + ln1_b[col]);
    }
  }
  // xp = LN1(x) @ Wp + bp   (f32 out + f16 LDS copy)
#pragma unroll
  for (int nt = 0; nt < 8; ++nt) {
    v8f c = {};
#pragma unroll
    for (int kt = 0; kt < 4; ++kt)
      c = WMMA_F16(ldsA(aL, 128, kt * 32, lane), gBt(wpt, 128, kt * 32, nt * 16, lane), c);
#pragma unroll
    for (int r = 0; r < 8; ++r) {
      const int rl = r + 8 * half, col = nt * 16 + cl;
      const float val = c[r] + bp[col];
      xp[(size_t)(row0 + rl) * 128 + col] = val;
      xpL[rl * 128 + col] = (_Float16)val;
    }
  }
  // Q (scale 1/sqrt(DH)=0.25 folded), K, V(transposed [B,H,16,N])
#pragma unroll
  for (int nt = 0; nt < 8; ++nt) {
    v8f cq = {}, ck = {}, cv = {};
#pragma unroll
    for (int kt = 0; kt < 4; ++kt) {
      v16h A = ldsA(xpL, 128, kt * 32, lane);
      cq = WMMA_F16(A, gBt(wqt, 128, kt * 32, nt * 16, lane), cq);
      ck = WMMA_F16(A, gBt(wkt, 128, kt * 32, nt * 16, lane), ck);
      cv = WMMA_F16(A, gBt(wvt, 128, kt * 32, nt * 16, lane), cv);
    }
#pragma unroll
    for (int r = 0; r < 8; ++r) {
      const int rl = r + 8 * half, col = nt * 16 + cl;
      const int flat = row0 + rl;
      qh[(size_t)flat * 128 + col] = (_Float16)(0.25f * cq[r]);
      kh[(size_t)flat * 128 + col] = (_Float16)ck[r];
      const int bb = flat >> 9, nn = flat & 511;           // h = nt, d = cl
      vth[((size_t)(bb * Hdim + nt) * 16 + cl) * Ndim + nn] = (_Float16)cv[r];
    }
  }
}

// ===== kernel 2: fused WMMA edge-MLP + masked attention + PV + LN2 + FFN =====
// grid: 256 blocks = (b 0..7) x (n-tile 0..31); block: 256 thr = 8 waves = 8 heads
__global__ __launch_bounds__(256) void attn_ffn_kernel(
    const float* __restrict__ xp, const _Float16* __restrict__ qh,
    const _Float16* __restrict__ kh, const _Float16* __restrict__ vth,
    const int* __restrict__ adj, const float* __restrict__ edge_attr,
    const float* __restrict__ eln_g, const float* __restrict__ eln_b,
    const _Float16* __restrict__ we1t, const _Float16* __restrict__ we2t,
    const float* __restrict__ be1, const float* __restrict__ be2,
    const float* __restrict__ ln2_g, const float* __restrict__ ln2_b,
    const _Float16* __restrict__ w1t, const float* __restrict__ ffb1,
    const _Float16* __restrict__ w2t, const float* __restrict__ ffb2,
    float* __restrict__ out) {
  __shared__ __align__(32) _Float16 eLN[256 * 16];   // LN'd edges (wave-private rows)
  __shared__ __align__(32) _Float16 hid[256 * 32];   // relu hidden (wave-private rows)
  __shared__ float    ebL[8][16][16];                // tanh*5 bias per head (cross-wave)
  __shared__ __align__(32) _Float16 pL[8][16][16];   // P staging (wave-private)
  __shared__ __align__(32) _Float16 sW1t[32 * 16];   // We1 transposed [n][k]
  __shared__ __align__(32) _Float16 sW2t[16 * 32];   // We2 padded transposed [n][k]
  __shared__ unsigned char sAdj[256];
  __shared__ float sbe1[32], sbe2[8], sg[16], sb[16];
  __shared__ float    hL[16 * 128];                  // h tile f32 (residual source)
  __shared__ __align__(32) _Float16 aL2[16 * 128];   // LN2(h) f16
  __shared__ __align__(32) _Float16 t1L[16 * 256];   // relu(ffW1) f16

  const int tid = threadIdx.x;
  const int b = blockIdx.x >> 5;
  const int n0 = (blockIdx.x & 31) * 16;
  const int h = tid >> 5;
  const int lane = tid & 31;
  const int half = lane >> 4, cl = lane & 15;

  sW1t[tid] = we1t[tid];  sW1t[tid + 256] = we1t[tid + 256];
  sW2t[tid] = we2t[tid];  sW2t[tid + 256] = we2t[tid + 256];
  if (tid < 32) sbe1[tid] = be1[tid];
  if (tid < 8)  sbe2[tid] = be2[tid];
  if (tid < 16) { sg[tid] = eln_g[tid]; sb[tid] = eln_b[tid]; }
  __syncthreads();

  // Q operand (K=16 zero-padded; scale pre-folded in kernel1)
  v16h qa;
  {
    const _Float16* qp = qh + (size_t)(b * Ndim + n0 + cl) * 128 + h * 16;
    v16h qv = *(const v16h*)(qp);
#pragma unroll
    for (int e = 0; e < 16; ++e)
      qa[e] = (half == 0) ? qv[e] : (_Float16)0.f;
  }

  float m_run[8], l_run[8];
  v8f ctx = {};
#pragma unroll
  for (int r = 0; r < 8; ++r) { m_run[r] = -3.0e38f; l_run[r] = 0.f; }

  const int nlid = tid >> 4, mlid = tid & 15;
  const int rt0 = 2 * h;   // this wave's pair of 16-row edge tiles
  const v16h z16 = {};

  // ---- streamed inputs: register double-buffer + non-temporal loads (read-once data)
  const int ng = n0 + nlid;
  const float* eaBase = edge_attr + ((size_t)(b * Ndim + ng) * Ndim + mlid) * 16;
  const int*   adBase = adj + (size_t)(b * Ndim + ng) * Ndim + mlid;
  float ev[16];
  {
    const v4f* p = (const v4f*)eaBase;
#pragma unroll
    for (int c = 0; c < 4; ++c) {
      v4f t = __builtin_nontemporal_load(p + c);
#pragma unroll
      for (int j = 0; j < 4; ++j) ev[c * 4 + j] = t[j];
    }
  }
  int adjv = __builtin_nontemporal_load(adBase);

  for (int mt = 0; mt < 32; ++mt) {
    const int m0 = mt * 16;
    // issue next tile's streamed loads now; consumed only at the rotate below
    float evn[16]; int adjn = 0;
    if (mt + 1 < 32) {
      const v4f* p = (const v4f*)(eaBase + (size_t)(mt + 1) * 256);
#pragma unroll
      for (int c = 0; c < 4; ++c) {
        v4f t = __builtin_nontemporal_load(p + c);
#pragma unroll
        for (int j = 0; j < 4; ++j) evn[c * 4 + j] = t[j];
      }
      adjn = __builtin_nontemporal_load(adBase + (mt + 1) * 16);
    }

    // ---- edge LN (one edge per thread) -> f16 LDS (wave-private rows); adj tile
    {
      sAdj[tid] = ((adjv != 0) || (ng == m0 + mlid)) ? 1 : 0;
      float s = 0.f, s2 = 0.f;
#pragma unroll
      for (int i = 0; i < 16; ++i) { s += ev[i]; s2 += ev[i] * ev[i]; }
      const float mean = s * (1.f / 16.f);
      const float var  = s2 * (1.f / 16.f) - mean * mean;
      const float rinv = rsqrtf(var + 1e-5f);
#pragma unroll
      for (int i = 0; i < 16; ++i)
        eLN[tid * 16 + i] = (_Float16)((ev[i] - mean) * rinv * sg[i] + sb[i]);
    }
    // (no barrier: eLN rows 32h..32h+31 produced & consumed by the same wave)

    // ---- edge-MLP layer 1 via WMMA: [256,16] @ [16,32], relu -> hid (same-wave)
#pragma unroll
    for (int rr = 0; rr < 2; ++rr) {
      const int rt = rt0 + rr;
      v16h A = ldsA_pad16(eLN + rt * 16 * 16, 16, lane);
#pragma unroll
      for (int ct = 0; ct < 2; ++ct) {
        v16h Bv = (half == 0) ? *(const v16h*)(sW1t + (ct * 16 + cl) * 16) : z16;
        v8f c;
#pragma unroll
        for (int r = 0; r < 8; ++r) c[r] = sbe1[ct * 16 + cl];
        c = WMMA_F16(A, Bv, c);
#pragma unroll
        for (int r = 0; r < 8; ++r)
          hid[(rt * 16 + r + 8 * half) * 32 + ct * 16 + cl] = (_Float16)fmaxf(c[r], 0.f);
      }
    }
    // (no barrier: hid rows for rt are same-wave)

    // ---- edge-MLP layer 2 via WMMA: [256,32] @ [32,16(pad)], tanh*5 -> ebL
#pragma unroll
    for (int rr = 0; rr < 2; ++rr) {
      const int rt = rt0 + rr;
      v16h Bv = *(const v16h*)(sW2t + cl * 32 + 16 * half);
      v8f c;
#pragma unroll
      for (int r = 0; r < 8; ++r) c[r] = (cl < 8) ? sbe2[cl] : 0.f;
      c = WMMA_F16(ldsA(hid + rt * 16 * 32, 32, 0, lane), Bv, c);
#pragma unroll
      for (int r = 0; r < 8; ++r) {
        const int ml = r + 8 * half;                  // edge = rt*16 + ml -> (nl=rt, ml)
        if (cl < 8) ebL[cl][rt][ml] = 5.f * tanhf(c[r]);
      }
    }
    __syncthreads();   // ebL + sAdj now visible to all waves

    // ---- scores: S = (Q/sqrt(DH)) @ K^T  (one 32B f16 load per lane)
    v16h kb = (half == 0)
                  ? *(const v16h*)(kh + (size_t)(b * Ndim + m0 + cl) * 128 + h * 16)
                  : z16;
    v8f zero = {};
    v8f sc = WMMA_F16(qa, kb, zero);

    // bias + leakyrelu + mask + online softmax
#pragma unroll
    for (int r = 0; r < 8; ++r) {
      const int nl = r + 8 * half;
      float vv = sc[r] + ebL[h][nl][cl];
      vv = (vv >= 0.f) ? vv : 0.2f * vv;
      const bool am = sAdj[nl * 16 + cl] != 0;
      const float sv = am ? vv : -3.0e38f;
      const float tmax = half_max16(sv);
      const float mnew = fmaxf(m_run[r], tmax);
      const float fac  = __expf(m_run[r] - mnew);
      const float p    = am ? __expf(vv - mnew) : 0.f;
      const float psum = half_sum16(p);
      l_run[r] = l_run[r] * fac + psum;
      ctx[r]   = ctx[r] * fac;
      m_run[r] = mnew;
      pL[h][nl][cl] = (_Float16)p;
    }

    // ---- PV: ctx += P @ V_tile  (V transposed -> one 32B f16 load per lane)
    v16h pa;
#pragma unroll
    for (int e = 0; e < 16; ++e)
      pa[e] = (half == 0) ? pL[h][cl][e] : (_Float16)0.f;
    v16h vb = (half == 0)
                  ? *(const v16h*)(vth + ((size_t)(b * Hdim + h) * 16 + cl) * Ndim + m0)
                  : z16;
    ctx = WMMA_F16(pa, vb, ctx);
    __syncthreads();   // protect ebL/sAdj overwrite next iteration

    // rotate streamed double-buffer
#pragma unroll
    for (int i = 0; i < 16; ++i) ev[i] = evn[i];
    adjv = adjn;
  }

  // ---- h tile = xp + ctx/l  -> LDS (f32, residual source)
#pragma unroll
  for (int r = 0; r < 8; ++r) {
    const int nl = r + 8 * half;
    const int col = h * 16 + cl;
    hL[nl * 128 + col] = xp[(size_t)(b * Ndim + n0 + nl) * 128 + col] + ctx[r] / l_run[r];
  }
  __syncthreads();

  // ---- LN2: row = tid>>4 handled by 16 lanes x 8 cols each
  {
    const int row = tid >> 4, il = tid & 15;
    float vals[8], s = 0.f, s2 = 0.f;
#pragma unroll
    for (int c = 0; c < 8; ++c) { float t = hL[row * 128 + il + 16 * c]; vals[c] = t; s += t; s2 += t * t; }
    s = half_sum16(s); s2 = half_sum16(s2);
    const float mean = s * (1.f / 128.f);
    const float var  = s2 * (1.f / 128.f) - mean * mean;
    const float rinv = rsqrtf(var + 1e-5f);
#pragma unroll
    for (int c = 0; c < 8; ++c) {
      const int col = il + 16 * c;
      aL2[row * 128 + col] = (_Float16)((vals[c] - mean) * rinv * ln2_g[col] + ln2_b[col]);
    }
  }
  __syncthreads();

  // ---- FFN layer 1: wave w -> col tiles {2w, 2w+1}; t1 = relu(aL2 @ ffW1 + ffb1)
#pragma unroll
  for (int q2 = 0; q2 < 2; ++q2) {
    const int nt = 2 * h + q2;
    v8f c = {};
#pragma unroll
    for (int kt = 0; kt < 4; ++kt)
      c = WMMA_F16(ldsA(aL2, 128, kt * 32, lane), gBt(w1t, 128, kt * 32, nt * 16, lane), c);
#pragma unroll
    for (int r = 0; r < 8; ++r) {
      const int rl = r + 8 * half, col = nt * 16 + cl;
      t1L[rl * 256 + col] = (_Float16)fmaxf(c[r] + ffb1[col], 0.f);
    }
  }
  __syncthreads();

  // ---- FFN layer 2: wave w -> col tile w; out = h + t1 @ ffW2 + ffb2
  {
    const int nt = h;
    v8f c = {};
#pragma unroll
    for (int kt = 0; kt < 8; ++kt)
      c = WMMA_F16(ldsA(t1L, 256, kt * 32, lane), gBt(w2t, 256, kt * 32, nt * 16, lane), c);
#pragma unroll
    for (int r = 0; r < 8; ++r) {
      const int rl = r + 8 * half, col = nt * 16 + cl;
      out[(size_t)(b * Ndim + n0 + rl) * 128 + col] = hL[rl * 128 + col] + c[r] + ffb2[col];
    }
  }
}

extern "C" void kernel_launch(void* const* d_in, const int* in_sizes, int n_in,
                              void* d_out, int out_size, void* d_ws, size_t ws_size,
                              hipStream_t stream) {
  const float* x         = (const float*)d_in[0];
  const int*   adj       = (const int*)  d_in[1];
  const float* edge_attr = (const float*)d_in[2];
  const float* ln1_g     = (const float*)d_in[3];
  const float* ln1_b     = (const float*)d_in[4];
  const float* Wp        = (const float*)d_in[5];
  const float* bp        = (const float*)d_in[6];
  const float* eln_g     = (const float*)d_in[7];
  const float* eln_b     = (const float*)d_in[8];
  const float* We1       = (const float*)d_in[9];
  const float* be1       = (const float*)d_in[10];
  const float* We2       = (const float*)d_in[11];
  const float* be2       = (const float*)d_in[12];
  const float* Wq        = (const float*)d_in[13];
  const float* Wk        = (const float*)d_in[14];
  const float* Wv        = (const float*)d_in[15];
  const float* ffW1      = (const float*)d_in[16];
  const float* ffb1      = (const float*)d_in[17];
  const float* ffW2      = (const float*)d_in[18];
  const float* ffb2      = (const float*)d_in[19];
  const float* ln2_g     = (const float*)d_in[20];
  const float* ln2_b     = (const float*)d_in[21];

  char* wsb = (char*)d_ws;
  float*    xp  = (float*)wsb;                            // 2 MB
  _Float16* qh  = (_Float16*)(wsb + (2u << 20));          // 1 MB
  _Float16* kh  = (_Float16*)(wsb + (3u << 20));          // 1 MB
  _Float16* vth = (_Float16*)(wsb + (4u << 20));          // 1 MB
  _Float16* wpt = (_Float16*)(wsb + (5u << 20));
  _Float16* wqt = wpt + 16384;
  _Float16* wkt = wqt + 16384;
  _Float16* wvt = wkt + 16384;
  _Float16* w1t = wvt + 16384;
  _Float16* w2t = w1t + 32768;
  _Float16* we1t = w2t + 32768;
  _Float16* we2t = we1t + 512;

  prep_kernel<<<128, 256, 0, stream>>>(Wp, Wq, Wk, Wv, ffW1, ffW2, We1, We2,
                                       wpt, wqt, wkt, wvt, w1t, w2t, we1t, we2t);
  const int tiles = (Bdim * Ndim) / 16;  // 256
  node_proj_kernel<<<tiles, 32, 0, stream>>>(x, ln1_g, ln1_b, bp, wpt, wqt, wkt, wvt,
                                             xp, qh, kh, vth);
  attn_ffn_kernel<<<Bdim * (Ndim / 16), 256, 0, stream>>>(
      xp, qh, kh, vth, adj, edge_attr, eln_g, eln_b, we1t, we2t, be1, be2,
      ln2_g, ln2_b, w1t, ffb1, w2t, ffb2, (float*)d_out);
}
